// GraphEmbedding_73641509257510
// MI455X (gfx1250) — matrix operand order repaired and checked by reading, not compiled, not run on hardware
//
#include <hip/hip_runtime.h>
#include <hip/hip_bf16.h>
#include <cmath>

// ---- problem constants (from reference) ----
#define DD    172          // feature dim D
#define D3    516          // 3*D
#define KP    544          // padded K dim (17 * 32)
#define NPG   192          // padded N dim (12 * 16)
#define NTG   4            // n-tiles per wave
#define NGRP  3            // n-groups (NPG / (NTG*16))
#define KN    20           // neighbors per node (K)
#define BB    1024         // batch
#define M2    (BB*KN)      // 20480 hop-2 rows

typedef __bf16 v16bf __attribute__((ext_vector_type(16)));
typedef __bf16 v8bf  __attribute__((ext_vector_type(8)));
typedef float  v8f   __attribute__((ext_vector_type(8)));

// ---------------------------------------------------------------------------
// Weight conversion: W [516,172] f32 (row-major, K x N) -> Wt [192,544] bf16
// (transposed + zero padded) so B fragments are contiguous per lane.
// ---------------------------------------------------------------------------
__global__ void k_cvt_w(const float* __restrict__ W, __bf16* __restrict__ Wt) {
    int idx = blockIdx.x * blockDim.x + threadIdx.x;
    if (idx >= NPG * KP) return;
    int n = idx / KP, k = idx % KP;
    float v = (n < DD && k < D3) ? W[(size_t)k * DD + n] : 0.0f;
    Wt[idx] = (__bf16)v;
}

// ---------------------------------------------------------------------------
// Hop-2 gather + K-sum:  S2[i] = [ sum_k h0(n2), sum_k cos(dt*w+b), sum_k ef ]
// one block per row i (of 20480), thread d = feature dim.
// ---------------------------------------------------------------------------
__global__ void k_build_S2(const float* __restrict__ mem, const float* __restrict__ nodef,
                           const float* __restrict__ edgef,
                           const float* __restrict__ time_w, const float* __restrict__ time_b,
                           const float* __restrict__ timestamps, const float* __restrict__ et2,
                           const int* __restrict__ n2, const int* __restrict__ e2,
                           __bf16* __restrict__ S2) {
    int i = blockIdx.x;
    int d = threadIdx.x;
    __bf16* out = S2 + (size_t)i * KP;
    for (int p = d; p < KP - D3; p += blockDim.x) out[D3 + p] = (__bf16)0.0f;   // K padding
    if (d >= DD) return;
    float ts = timestamps[i / KN];
    float w = time_w[d], b = time_b[d];
    float s0 = 0.f, s1 = 0.f, s2 = 0.f;
    for (int k = 0; k < KN; ++k) {
        int nn = n2[i * KN + k];
        int ee = e2[i * KN + k];
        s0 += mem[(size_t)nn * DD + d] + nodef[(size_t)nn * DD + d];
        float dt = ts - et2[i * KN + k];
        s1 += cosf(dt * w + b);
        s2 += edgef[(size_t)ee * DD + d];
    }
    out[d]          = (__bf16)s0;
    out[DD + d]     = (__bf16)s1;
    out[2 * DD + d] = (__bf16)s2;
}

// SF2 tail: columns [172,516) = [h0(neighbors_l1 flat), cos(time_b)], + pad.
__global__ void k_tail_SF2(const float* __restrict__ mem, const float* __restrict__ nodef,
                           const float* __restrict__ time_b, const int* __restrict__ n1,
                           __bf16* __restrict__ SF2) {
    int i = blockIdx.x;
    int d = threadIdx.x;
    __bf16* out = SF2 + (size_t)i * KP;
    for (int p = d; p < KP - D3; p += blockDim.x) out[D3 + p] = (__bf16)0.0f;
    if (d >= DD) return;
    int nn = n1[i];
    out[DD + d]     = (__bf16)(mem[(size_t)nn * DD + d] + nodef[(size_t)nn * DD + d]);
    out[2 * DD + d] = (__bf16)cosf(time_b[d]);
}

// Hop-1 gather + K-sum over h1 / time enc / edge features.
__global__ void k_build_S1(const float* __restrict__ h1, const float* __restrict__ edgef,
                           const float* __restrict__ time_w, const float* __restrict__ time_b,
                           const float* __restrict__ timestamps, const float* __restrict__ et1,
                           const int* __restrict__ e1, __bf16* __restrict__ S1) {
    int i = blockIdx.x;
    int d = threadIdx.x;
    __bf16* out = S1 + (size_t)i * KP;
    for (int p = d; p < KP - D3; p += blockDim.x) out[D3 + p] = (__bf16)0.0f;
    if (d >= DD) return;
    float ts = timestamps[i];
    float w = time_w[d], b = time_b[d];
    float s0 = 0.f, s1 = 0.f, s2 = 0.f;
    for (int k = 0; k < KN; ++k) {
        s0 += h1[((size_t)i * KN + k) * DD + d];
        float dt = ts - et1[i * KN + k];
        s1 += cosf(dt * w + b);
        s2 += edgef[(size_t)e1[i * KN + k] * DD + d];
    }
    out[d]          = (__bf16)s0;
    out[DD + d]     = (__bf16)s1;
    out[2 * DD + d] = (__bf16)s2;
}

__global__ void k_tail_SF1(const float* __restrict__ mem, const float* __restrict__ nodef,
                           const float* __restrict__ time_b, const int* __restrict__ src,
                           __bf16* __restrict__ SF1) {
    int i = blockIdx.x;
    int d = threadIdx.x;
    __bf16* out = SF1 + (size_t)i * KP;
    for (int p = d; p < KP - D3; p += blockDim.x) out[D3 + p] = (__bf16)0.0f;
    if (d >= DD) return;
    int nn = src[i];
    out[DD + d]     = (__bf16)(mem[(size_t)nn * DD + d] + nodef[(size_t)nn * DD + d]);
    out[2 * DD + d] = (__bf16)cosf(time_b[d]);
}

// ---------------------------------------------------------------------------
// WMMA GEMM:  Out[M,172] = act( A[M,544] @ Wt^T + biasScale*bias )
//   A bf16 row-major (stride KP), Wt bf16 [192][544] (pre-transposed weights).
//   One wave = one 16-row M-tile x one group of 4 n-tiles (32 accum VGPRs,
//   no spills, good occupancy). 3 groups cover N; A re-read 3x (cheap).
//   OUT_BF16: write bf16 at row-stride `outStride` (into next stage's A);
//   else fp32. RELU for GEMM1 stages.
// ---------------------------------------------------------------------------
template <bool OUT_BF16, bool RELU>
__global__ __launch_bounds__(128)
void k_wmma_gemm(const __bf16* __restrict__ A, const __bf16* __restrict__ Wt,
                 const float* __restrict__ bias, float biasScale,
                 void* __restrict__ Out, int outStride, int M) {
    int wave = blockIdx.x * (blockDim.x >> 5) + (threadIdx.x >> 5);
    int mt = wave / NGRP;            // m-tile index
    int g  = wave % NGRP;            // n-group index
    int m0 = mt * 16;
    if (m0 >= M) return;
    int n0 = g * (NTG * 16);
    int lane  = threadIdx.x & 31;
    int lhalf = lane >> 4;           // 0 | 1
    int lmod  = lane & 15;

    v8f acc[NTG];
#pragma unroll
    for (int t = 0; t < NTG; ++t) acc[t] = (v8f){0.f, 0.f, 0.f, 0.f, 0.f, 0.f, 0.f, 0.f};

    // A fragment base for this lane: row m0+lmod, K-subgroup per lane half.
    const __bf16* arow = A + (size_t)(m0 + lmod) * KP + lhalf * 8;
    const __bf16* brow = Wt + (size_t)(n0 + lmod) * KP + lhalf * 8;

    for (int kt = 0; kt < KP / 32; ++kt) {
        const __bf16* ap = arow + kt * 32;
        v8bf alo = *(const v8bf*)ap;            // K {0..7} or {8..15}
        v8bf ahi = *(const v8bf*)(ap + 16);     // K {16..23} or {24..31}
        v16bf afrag = __builtin_shufflevector(alo, ahi,
            0, 1, 2, 3, 4, 5, 6, 7, 8, 9, 10, 11, 12, 13, 14, 15);
#pragma unroll
        for (int j = 0; j < NTG; ++j) {
            const __bf16* bp = brow + (size_t)(j * 16) * KP + kt * 32;
            v8bf blo = *(const v8bf*)bp;
            v8bf bhi = *(const v8bf*)(bp + 16);
            v16bf bfrag = __builtin_shufflevector(blo, bhi,
                0, 1, 2, 3, 4, 5, 6, 7, 8, 9, 10, 11, 12, 13, 14, 15);
            acc[j] = __builtin_amdgcn_wmma_f32_16x16x32_bf16(
                false, afrag, false, bfrag, (short)0, acc[j], false, false);
        }
    }

    // Epilogue: bias (+optional ReLU), skip padded n >= 172.
#pragma unroll
    for (int j = 0; j < NTG; ++j) {
        int n = n0 + j * 16 + lmod;
        if (n >= DD) continue;
        float bv = biasScale * bias[n];
#pragma unroll
        for (int r = 0; r < 8; ++r) {
            int row = m0 + r + lhalf * 8;
            float v = acc[j][r] + bv;
            if (RELU) v = v > 0.f ? v : 0.f;
            if (OUT_BF16)
                ((__bf16*)Out)[(size_t)row * outStride + n] = (__bf16)v;
            else
                ((float*)Out)[(size_t)row * outStride + n] = v;
        }
    }
}

// ---------------------------------------------------------------------------
extern "C" void kernel_launch(void* const* d_in, const int* in_sizes, int n_in,
                              void* d_out, int out_size, void* d_ws, size_t ws_size,
                              hipStream_t stream) {
    (void)in_sizes; (void)n_in; (void)out_size; (void)ws_size;
    const float* mem     = (const float*)d_in[0];
    const float* nodef   = (const float*)d_in[1];
    const float* edgef   = (const float*)d_in[2];
    const float* time_w  = (const float*)d_in[3];
    const float* time_b  = (const float*)d_in[4];
    const float* W1      = (const float*)d_in[5];   // [2,516,172]
    const float* b1      = (const float*)d_in[6];   // [2,172]
    const float* W2      = (const float*)d_in[7];
    const float* b2      = (const float*)d_in[8];
    const float* tstamps = (const float*)d_in[9];
    const float* et1     = (const float*)d_in[10];
    const float* et2     = (const float*)d_in[11];
    const int*   src     = (const int*)d_in[12];
    const int*   n1      = (const int*)d_in[13];
    const int*   e1      = (const int*)d_in[14];
    const int*   n2      = (const int*)d_in[15];
    const int*   e2      = (const int*)d_in[16];

    // workspace carve-up (256B aligned)
    char* ws = (char*)d_ws;
    size_t off = 0;
    auto take = [&](size_t bytes) { char* p = ws + off; off = (off + bytes + 255) & ~(size_t)255; return p; };
    __bf16* S2   = (__bf16*)take((size_t)M2 * KP * 2);
    __bf16* SF2  = (__bf16*)take((size_t)M2 * KP * 2);
    float*  h1   = (float*) take((size_t)M2 * DD * 4);
    __bf16* S1   = (__bf16*)take((size_t)BB * KP * 2);
    __bf16* SF1  = (__bf16*)take((size_t)BB * KP * 2);
    __bf16* Wt1a = (__bf16*)take((size_t)NPG * KP * 2);
    __bf16* Wt1b = (__bf16*)take((size_t)NPG * KP * 2);
    __bf16* Wt2a = (__bf16*)take((size_t)NPG * KP * 2);
    __bf16* Wt2b = (__bf16*)take((size_t)NPG * KP * 2);

    // 1) weights -> transposed/padded bf16
    {
        int nblk = (NPG * KP + 255) / 256;
        k_cvt_w<<<nblk, 256, 0, stream>>>(W1,                    Wt1a);
        k_cvt_w<<<nblk, 256, 0, stream>>>(W1 + (size_t)D3 * DD,  Wt1b);
        k_cvt_w<<<nblk, 256, 0, stream>>>(W2,                    Wt2a);
        k_cvt_w<<<nblk, 256, 0, stream>>>(W2 + (size_t)D3 * DD,  Wt2b);
    }

    // 2) hop-2 gather/sum + SF2 tail segments
    k_build_S2<<<M2, 192, 0, stream>>>(mem, nodef, edgef, time_w, time_b,
                                       tstamps, et2, n2, e2, S2);
    k_tail_SF2<<<M2, 192, 0, stream>>>(mem, nodef, time_b, n1, SF2);

    // 3) hop-2 GEMM1 (relu, bias*K) -> SF2[:, 0:172) as bf16
    {
        int waves = (M2 / 16) * NGRP;               // 1280 * 3
        k_wmma_gemm<true, true><<<waves / 4, 128, 0, stream>>>(
            S2, Wt1a, b1, (float)KN, SF2, KP, M2);
        // 4) hop-2 GEMM2 -> h1 fp32 [20480,172]
        k_wmma_gemm<false, false><<<waves / 4, 128, 0, stream>>>(
            SF2, Wt2a, b2, 1.0f, h1, DD, M2);
    }

    // 5) hop-1 gather/sum + SF1 tail
    k_build_S1<<<BB, 192, 0, stream>>>(h1, edgef, time_w, time_b, tstamps, et1, e1, S1);
    k_tail_SF1<<<BB, 192, 0, stream>>>(mem, nodef, time_b, src, SF1);

    // 6) hop-1 GEMM1 -> SF1[:, 0:172)
    {
        int waves = (BB / 16) * NGRP;               // 64 * 3
        k_wmma_gemm<true, true><<<waves / 4, 128, 0, stream>>>(
            S1, Wt1b, b1 + DD, (float)KN, SF1, KP, BB);
        // 7) hop-1 GEMM2 -> d_out fp32 [1024,172]
        k_wmma_gemm<false, false><<<waves / 4, 128, 0, stream>>>(
            SF1, Wt2b, b2 + DD, 1.0f, d_out, DD, BB);
    }
}
